// SalientBranch_32177894981821
// MI455X (gfx1250) — compile-verified, matrix operand order checked by reading, not run on hardware
//
#include <hip/hip_runtime.h>
#include <hip/hip_bf16.h>
#include <hip/hip_fp16.h>

typedef _Float16 h16;
typedef __attribute__((ext_vector_type(16))) _Float16 v16h;
typedef __attribute__((ext_vector_type(8)))  _Float16 v8h;
typedef __attribute__((ext_vector_type(8)))  float    v8f;
typedef __attribute__((ext_vector_type(4)))  unsigned int u32x4;
typedef __attribute__((ext_vector_type(8)))  int      i32x8;
typedef __attribute__((ext_vector_type(4)))  int      i32x4;

#define SHUF16(a, b) __builtin_shufflevector(a, b, 0, 1, 2, 3, 4, 5, 6, 7, 8, 9, 10, 11, 12, 13, 14, 15)

__device__ __forceinline__ float gelu_f(float x) {
    return 0.5f * x * (1.0f + erff(x * 0.70710678118f));
}

// ---------------------------------------------------------------------------
// Tensor Data Mover: 2D tile global -> LDS (f16 elements).
// Compile is a hard probe: device pass errors out if the builtin is missing,
// so a clean build proves tensor_load_to_lds/s_wait_tensorcnt are emitted.
// ---------------------------------------------------------------------------
#if defined(__HIP_DEVICE_COMPILE__)
#if !__has_builtin(__builtin_amdgcn_tensor_load_to_lds)
#error "gfx1250 TDM builtin __builtin_amdgcn_tensor_load_to_lds not available"
#endif
#define USE_TDM 1
#else
#define USE_TDM 0
#endif

#if USE_TDM
__device__ __forceinline__ void tdm_load_2d(unsigned lds_off, const void* gsrc,
                                            unsigned tile_d0, unsigned tile_d1,
                                            unsigned tensor_d0, unsigned tensor_d1,
                                            unsigned stride0) {
    unsigned long long ga = (unsigned long long)gsrc;
    u32x4 g0;
    g0[0] = 1u;                                                // count=1 user descriptor
    g0[1] = lds_off;                                           // lds_addr
    g0[2] = (unsigned)ga;                                      // global_addr[31:0]
    g0[3] = ((unsigned)(ga >> 32) & 0x01FFFFFFu) | (2u << 30); // addr[56:32] | type=2
    i32x8 g1;
    g1[0] = 0x00010000;                                        // mask=0, data_size=2B
    g1[1] = (int)((tensor_d0 & 0xFFFFu) << 16);                                   // dim0[15:0]
    g1[2] = (int)(((tensor_d0 >> 16) & 0xFFFFu) | ((tensor_d1 & 0xFFFFu) << 16)); // dim0 hi | dim1 lo
    g1[3] = (int)(((tensor_d1 >> 16) & 0xFFFFu) | ((tile_d0 & 0xFFFFu) << 16));   // dim1 hi | tile_dim0
    g1[4] = (int)(tile_d1 & 0xFFFFu);                          // tile_dim1 | tile_dim2=0
    g1[5] = (int)stride0;                                      // tensor_dim0_stride[31:0]
    g1[6] = 0;
    g1[7] = 0;
    i32x4 z4 = {0, 0, 0, 0};
#if __clang_major__ >= 23
    i32x8 z8 = {0, 0, 0, 0, 0, 0, 0, 0};
    __builtin_amdgcn_tensor_load_to_lds(g0, g1, z4, z4, z8, 0);
#else
    __builtin_amdgcn_tensor_load_to_lds(g0, g1, z4, z4, 0);
#endif
}
#endif

// ---------------------------------------------------------------------------
// B operands (weights / k^T / v) live in K-panel layout:
//   element (k, n)  ->  [k/32][n][k%32]   (32 contiguous f16 per (panel, n))
// so a K-chunk panel is a 2D tile {tile_d0=32 contig, tile_d1=64 rows, stride 32}
// for the TDM, and each wave's B fragment is two ds_load_b128.
// ---------------------------------------------------------------------------

// Implicit-GEMM 3x3 conv (pad=1) + folded BN + ReLU. NHWC f16.
// Block = 4 waves -> 32(M) x 64(N) C-tile, 2 WMMAs per wave per K-step.
__global__ __launch_bounds__(128) void k_conv3x3(
    const h16* __restrict__ in, const h16* __restrict__ wp,
    const float* __restrict__ scale, const float* __restrict__ bias,
    h16* __restrict__ out, int Nn, int H, int W, int Ci, int Co) {
    __shared__ h16 As[32 * 32];
    __shared__ h16 Bs[64 * 32];
    const int tid = threadIdx.x, wave = tid >> 5, lane = tid & 31;
    const int sub = lane >> 4, l16 = lane & 15;
    const int m0 = blockIdx.x * 32, n0b = blockIdx.y * 64;
    const int M = Nn * H * W, K = 9 * Ci;
    v8f acc0 = {}, acc1 = {};
    for (int k0 = 0; k0 < K; k0 += 32) {
        const int tap = k0 / Ci, cb = k0 % Ci;
        const int dy = tap / 3 - 1, dx = tap % 3 - 1;
        {   // A tile: 32 pixels x 32 k, 8 contiguous halves per thread (b128)
            int e = tid * 8, r = e >> 5, kk = e & 31;
            int m = m0 + r; if (m >= M) m = M - 1;
            int b = m / (H * W), rem = m % (H * W), y = rem / W, x = rem % W;
            int yy = y + dy, xx = x + dx;
            bool ok = (yy >= 0 && yy < H && xx >= 0 && xx < W);
            const v8h* src = (const v8h*)(in + (((size_t)(b * H + (ok ? yy : 0)) * W + (ok ? xx : 0)) * Ci + cb + kk));
            v8h z = {};
            *(v8h*)&As[r * 32 + kk] = ok ? *src : z;
        }
        const h16* pbase = wp + ((size_t)(k0 >> 5) * Co + n0b) * 32;
#if USE_TDM
        if (wave == 0) {
            unsigned rows = (unsigned)(Co - n0b); if (rows > 64u) rows = 64u;
            tdm_load_2d((unsigned)(unsigned long long)(const void*)&Bs[0],
                        (const void*)pbase, 32u, 64u, 32u, rows, 32u);
            __builtin_amdgcn_s_wait_tensorcnt(0);
        }
#else
        {
            int idx = tid * 16, col = n0b + (idx >> 5);
            v8h z = {};
            const v8h* s = (const v8h*)(pbase + idx);
            *(v8h*)&Bs[idx]     = (col < Co) ? s[0] : z;
            *(v8h*)&Bs[idx + 8] = (col < Co) ? s[1] : z;
        }
#endif
        __syncthreads();
        v8h a00 = *(const v8h*)&As[l16 * 32 + sub * 8];
        v8h a01 = *(const v8h*)&As[l16 * 32 + 16 + sub * 8];
        v8h a10 = *(const v8h*)&As[(16 + l16) * 32 + sub * 8];
        v8h a11 = *(const v8h*)&As[(16 + l16) * 32 + 16 + sub * 8];
        v8h b0  = *(const v8h*)&Bs[(wave * 16 + l16) * 32 + sub * 8];
        v8h b1  = *(const v8h*)&Bs[(wave * 16 + l16) * 32 + 16 + sub * 8];
        v16h af0 = SHUF16(a00, a01), af1 = SHUF16(a10, a11), bf = SHUF16(b0, b1);
        acc0 = __builtin_amdgcn_wmma_f32_16x16x32_f16(false, af0, false, bf, (short)0, acc0, false, false);
        acc1 = __builtin_amdgcn_wmma_f32_16x16x32_f16(false, af1, false, bf, (short)0, acc1, false, false);
        __syncthreads();
    }
    const int col = n0b + wave * 16 + l16;
#pragma unroll
    for (int r = 0; r < 8; ++r) {
        int row0 = m0 + sub * 8 + r, row1 = row0 + 16;
        if (col < Co) {
            if (row0 < M) {
                float v = acc0[r] * scale[col] + bias[col];
                out[(size_t)row0 * Co + col] = (h16)(v > 0.f ? v : 0.f);
            }
            if (row1 < M) {
                float v = acc1[r] * scale[col] + bias[col];
                out[(size_t)row1 * Co + col] = (h16)(v > 0.f ? v : 0.f);
            }
        }
    }
}

// Batched GEMM: A[M,K]f16 @ Bpanels[K/32][N][32]f16 (+bias) -> f16/f32.
// act: 0=none 1=relu 2=gelu(exact).
__global__ __launch_bounds__(128) void k_gemm(
    const h16* __restrict__ A, const h16* __restrict__ B,
    const float* __restrict__ bias, h16* __restrict__ outh, float* __restrict__ outf,
    int M, int K, int N, long long sA, long long sB, long long sC, int act) {
    __shared__ h16 As[32 * 32];
    __shared__ h16 Bs[64 * 32];
    const int tid = threadIdx.x, wave = tid >> 5, lane = tid & 31;
    const int sub = lane >> 4, l16 = lane & 15;
    const int m0 = blockIdx.x * 32, n0b = blockIdx.y * 64, z = blockIdx.z;
    const h16* Ab = A + (size_t)z * sA;
    const h16* Bb = B + (size_t)z * sB;
    v8f acc0 = {}, acc1 = {};
    for (int k0 = 0; k0 < K; k0 += 32) {
        {
            int e = tid * 8, r = e >> 5, kk = e & 31;
            int m = m0 + r; if (m >= M) m = M - 1;
            *(v8h*)&As[r * 32 + kk] = *(const v8h*)(Ab + (size_t)m * K + k0 + kk);
        }
        const h16* pbase = Bb + ((size_t)(k0 >> 5) * N + n0b) * 32;
#if USE_TDM
        if (wave == 0) {
            unsigned rows = (unsigned)(N - n0b); if (rows > 64u) rows = 64u;
            tdm_load_2d((unsigned)(unsigned long long)(const void*)&Bs[0],
                        (const void*)pbase, 32u, 64u, 32u, rows, 32u);
            __builtin_amdgcn_s_wait_tensorcnt(0);
        }
#else
        {
            int idx = tid * 16, col = n0b + (idx >> 5);
            v8h z = {};
            const v8h* s = (const v8h*)(pbase + idx);
            *(v8h*)&Bs[idx]     = (col < N) ? s[0] : z;
            *(v8h*)&Bs[idx + 8] = (col < N) ? s[1] : z;
        }
#endif
        __syncthreads();
        v8h a00 = *(const v8h*)&As[l16 * 32 + sub * 8];
        v8h a01 = *(const v8h*)&As[l16 * 32 + 16 + sub * 8];
        v8h a10 = *(const v8h*)&As[(16 + l16) * 32 + sub * 8];
        v8h a11 = *(const v8h*)&As[(16 + l16) * 32 + 16 + sub * 8];
        v8h b0  = *(const v8h*)&Bs[(wave * 16 + l16) * 32 + sub * 8];
        v8h b1  = *(const v8h*)&Bs[(wave * 16 + l16) * 32 + 16 + sub * 8];
        v16h af0 = SHUF16(a00, a01), af1 = SHUF16(a10, a11), bf = SHUF16(b0, b1);
        acc0 = __builtin_amdgcn_wmma_f32_16x16x32_f16(false, af0, false, bf, (short)0, acc0, false, false);
        acc1 = __builtin_amdgcn_wmma_f32_16x16x32_f16(false, af1, false, bf, (short)0, acc1, false, false);
        __syncthreads();
    }
    h16*   oh = outh ? outh + (size_t)z * sC : nullptr;
    float* of = outf ? outf + (size_t)z * sC : nullptr;
    const int col = n0b + wave * 16 + l16;
    if (col < N) {
#pragma unroll
        for (int r = 0; r < 8; ++r) {
            int row0 = m0 + sub * 8 + r, row1 = row0 + 16;
#pragma unroll
            for (int h = 0; h < 2; ++h) {
                int row = h ? row1 : row0;
                if (row < M) {
                    float v = h ? acc1[r] : acc0[r];
                    if (bias) v += bias[col];
                    if (act == 1) v = fmaxf(v, 0.f);
                    else if (act == 2) v = gelu_f(v);
                    size_t o = (size_t)row * N + col;
                    if (oh) oh[o] = (h16)v;
                    if (of) of[o] = v;
                }
            }
        }
    }
}

// -------------------------- pointwise / helper kernels ---------------------
__global__ void k_nchw2nhwc(const float* in, h16* out, int N, int C, int H, int W) {
    size_t i = (size_t)blockIdx.x * 256 + threadIdx.x;
    size_t total = (size_t)N * C * H * W;
    if (i >= total) return;
    int c = i % C; size_t r = i / C; int x = r % W; r /= W; int y = r % H; int b = r / H;
    out[i] = (h16)in[(((size_t)b * C + c) * H + y) * W + x];
}

// conv weight OIHW f32 -> K-panel f16: (k=tap*Ci+ci, co) -> [k/32][co][k%32]
__global__ void k_pack_conv(const float* w, h16* out, int Ci, int Co) {
    size_t i = (size_t)blockIdx.x * 256 + threadIdx.x;
    size_t total = (size_t)9 * Ci * Co;
    if (i >= total) return;
    int kk = i & 31; size_t r = i >> 5; int co = r % Co; int p = r / Co;
    int k = p * 32 + kk, tap = k / Ci, ci = k % Ci;
    out[i] = (h16)w[((size_t)co * Ci + ci) * 9 + tap];
}

// linear weight [K][N] f32 -> K-panel f16
__global__ void k_pack_lin(const float* w, h16* out, int K, int N) {
    size_t i = (size_t)blockIdx.x * 256 + threadIdx.x;
    size_t total = (size_t)K * N;
    if (i >= total) return;
    int kk = i & 31; size_t r = i >> 5; int n = r % N; int p = r / N;
    out[i] = (h16)w[(size_t)(p * 32 + kk) * N + n];
}

__global__ void k_bnfold(const float* b, const float* bb, const float* g,
                         const float* m, const float* v, float* sc, float* bi, int C) {
    int c = blockIdx.x * 256 + threadIdx.x;
    if (c >= C) return;
    float s = g[c] * rsqrtf(v[c] + 1e-5f);
    sc[c] = s;
    bi[c] = (b[c] - m[c]) * s + bb[c];
}

__global__ void k_up2(const h16* in, h16* out, int N, int H, int W, int C) {
    int Ho = 2 * H, Wo = 2 * W;
    size_t total = (size_t)N * Ho * Wo * C;
    size_t i = (size_t)blockIdx.x * 256 + threadIdx.x;
    if (i >= total) return;
    int c = i % C; size_t r = i / C; int x = r % Wo; r /= Wo; int y = r % Ho; int b = r / Ho;
    float py = y * ((float)(H - 1) / (float)(Ho - 1));   // align_corners=True
    float px = x * ((float)(W - 1) / (float)(Wo - 1));
    int y0 = (int)py, x0 = (int)px;
    int y1 = min(y0 + 1, H - 1), x1 = min(x0 + 1, W - 1);
    float fy = py - y0, fx = px - x0;
    auto g = [&](int yy, int xx) { return (float)in[(((size_t)b * H + yy) * W + xx) * C + c]; };
    out[i] = (h16)((1 - fy) * ((1 - fx) * g(y0, x0) + fx * g(y0, x1)) +
                   fy * ((1 - fx) * g(y1, x0) + fx * g(y1, x1)));
}

__global__ void k_mul(const h16* a, const h16* b, h16* o, size_t n) {
    size_t i = (size_t)blockIdx.x * 256 + threadIdx.x;
    if (i < n) o[i] = (h16)((float)a[i] * (float)b[i]);
}

__global__ void k_cat2(const h16* a, int Ca, const h16* b, int Cb, h16* o, int P) {
    int Ct = Ca + Cb;
    size_t total = (size_t)P * Ct;
    size_t i = (size_t)blockIdx.x * 256 + threadIdx.x;
    if (i >= total) return;
    int c = i % Ct; size_t p = i / Ct;
    o[i] = (c < Ca) ? a[p * Ca + c] : b[p * Cb + (c - Ca)];
}

__global__ void k_cat3(const h16* a, const h16* b, const h16* c, h16* o, int P, int C) {
    size_t total = (size_t)P * 3 * C;
    size_t i = (size_t)blockIdx.x * 256 + threadIdx.x;
    if (i >= total) return;
    int ch = i % (3 * C); size_t p = i / (3 * C);
    const h16* s = (ch < C) ? a : (ch < 2 * C) ? b : c;
    o[i] = s[p * C + (ch % C)];
}

__global__ void k_ln(const h16* x, const float* w, const float* b, h16* y, int P, int C) {
    int p = blockIdx.x * 256 + threadIdx.x;
    if (p >= P) return;
    const h16* xr = x + (size_t)p * C;
    float s = 0, s2 = 0;
    for (int c = 0; c < C; ++c) { float v = (float)xr[c]; s += v; s2 += v * v; }
    float mu = s / C, var = s2 / C - mu * mu;
    float inv = rsqrtf(var + 1e-6f);
    h16* yr = y + (size_t)p * C;
    for (int c = 0; c < C; ++c) yr[c] = (h16)(((float)xr[c] - mu) * inv * w[c] + b[c]);
}

__global__ void k_dwconv(const h16* in, const float* w, const float* b, h16* out,
                         int N, int H, int W, int C, int ks, int pad, int addres) {
    size_t total = (size_t)N * H * W * C;
    size_t i = (size_t)blockIdx.x * 256 + threadIdx.x;
    if (i >= total) return;
    int c = i % C; size_t r = i / C; int x = r % W; r /= W; int y = r % H; int bn = r / H;
    float acc = b[c];
    const float* wc = w + (size_t)c * ks * ks;
    for (int dy = 0; dy < ks; ++dy)
        for (int dx = 0; dx < ks; ++dx) {
            int yy = y + dy - pad, xx = x + dx - pad;
            if (yy >= 0 && yy < H && xx >= 0 && xx < W)
                acc += (float)in[(((size_t)bn * H + yy) * W + xx) * C + c] * wc[dy * ks + dx];
        }
    if (addres) acc += (float)in[i];
    out[i] = (h16)acc;
}

__global__ void k_gelu(const h16* a, h16* o, size_t n) {
    size_t i = (size_t)blockIdx.x * 256 + threadIdx.x;
    if (i < n) o[i] = (h16)gelu_f((float)a[i]);
}

__global__ void k_pool8(const h16* in, h16* out, int B_, int C) {   // 56x56 -> 7x7 mean
    size_t total = (size_t)B_ * 49 * C;
    size_t i = (size_t)blockIdx.x * 256 + threadIdx.x;
    if (i >= total) return;
    int c = i % C; size_t r = i / C; int px = r % 7; r /= 7; int py = r % 7; int b = r / 7;
    float s = 0;
    for (int dy = 0; dy < 8; ++dy)
        for (int dx = 0; dx < 8; ++dx)
            s += (float)in[(((size_t)b * 56 + py * 8 + dy) * 56 + px * 8 + dx) * C + c];
    out[i] = (h16)(s * (1.f / 64.f));
}

// kv [B,HW,512] -> kT (B operand, K=32 panels: [bh][HW][32])
//                  v  (B operand, K=HW panels: [bh][HW/32][32 n][32 k])
__global__ void k_split_kv(const h16* kv, h16* kT, h16* v, int B_) {
    size_t total = (size_t)B_ * 3136 * 512;
    size_t i = (size_t)blockIdx.x * 256 + threadIdx.x;
    if (i >= total) return;
    int c = i & 511; size_t r = i >> 9; int p = r % 3136; int b = r / 3136;
    int e = c >> 8, hh = (c >> 5) & 7, j = c & 31;
    int bh = b * 8 + hh;
    if (e == 0) kT[((size_t)bh * 3136 + p) * 32 + j] = kv[i];
    else        v[(size_t)bh * 100352 + ((size_t)(p >> 5) * 32 + j) * 32 + (p & 31)] = kv[i];
}

__global__ void k_build_m(const h16* pm, h16* mt, int B_) {     // [B,49,512] -> [B*8][98][32]*hd^-.5
    size_t total = (size_t)B_ * 49 * 512;
    size_t i = (size_t)blockIdx.x * 256 + threadIdx.x;
    if (i >= total) return;
    int c = i & 511; size_t r = i >> 9; int t49 = r % 49; int b = r / 49;
    int t = t49 * 2 + (c >> 8), hh = (c >> 5) & 7, j = c & 31;
    mt[((size_t)(b * 8 + hh) * 98 + t) * 32 + j] = (h16)((float)pm[i] * 0.17677669529f);
}

__global__ __launch_bounds__(256) void k_softmax(const float* x, h16* y, int N) {
    __shared__ float red[256];
    const int r = blockIdx.x, tid = threadIdx.x;
    const float* xr = x + (size_t)r * N;
    float mx = -1e30f;
    for (int i = tid; i < N; i += 256) mx = fmaxf(mx, xr[i]);
    red[tid] = mx; __syncthreads();
    for (int s = 128; s > 0; s >>= 1) { if (tid < s) red[tid] = fmaxf(red[tid], red[tid + s]); __syncthreads(); }
    mx = red[0]; __syncthreads();
    float sm = 0;
    for (int i = tid; i < N; i += 256) sm += expf(xr[i] - mx);
    red[tid] = sm; __syncthreads();
    for (int s = 128; s > 0; s >>= 1) { if (tid < s) red[tid] += red[tid + s]; __syncthreads(); }
    float inv = 1.f / red[0];
    h16* yr = y + (size_t)r * N;
    for (int i = tid; i < N; i += 256) yr[i] = (h16)(expf(xr[i] - mx) * inv);
}

__global__ void k_o_up(const h16* o, h16* out, int B_) {   // [B*8][98][32] -> NHWC 56x56x512, half-pixel
    size_t total = (size_t)B_ * 56 * 56 * 512;
    size_t i = (size_t)blockIdx.x * 256 + threadIdx.x;
    if (i >= total) return;
    int c = i & 511; size_t r = i >> 9; int x = r % 56; r /= 56; int y = r % 56; int b = r / 56;
    int head = c >> 6, half = (c >> 5) & 1, j = c & 31, bh = b * 8 + head;
    float py = (y + 0.5f) * 0.125f - 0.5f, px = (x + 0.5f) * 0.125f - 0.5f;
    float y0f = floorf(py), x0f = floorf(px);
    float fy = py - y0f, fx = px - x0f;
    int y0 = min(max((int)y0f, 0), 6), y1 = min(max((int)y0f + 1, 0), 6);
    int x0 = min(max((int)x0f, 0), 6), x1 = min(max((int)x0f + 1, 0), 6);
    auto g = [&](int yy, int xx) {
        int t = (yy * 7 + xx) * 2 + half;
        return (float)o[((size_t)bh * 98 + t) * 32 + j];
    };
    out[i] = (h16)((1 - fy) * ((1 - fx) * g(y0, x0) + fx * g(y0, x1)) +
                   fy * ((1 - fx) * g(y1, x0) + fx * g(y1, x1)));
}

__global__ void k_axpy(const h16* x, const h16* y, const float* ls, h16* o, int P, int C) {
    size_t total = (size_t)P * C;
    size_t i = (size_t)blockIdx.x * 256 + threadIdx.x;
    if (i >= total) return;
    o[i] = (h16)((float)x[i] + ls[i % C] * (float)y[i]);
}

__global__ void k_axpy_f32(const h16* x, const h16* y, const float* ls, float* o, int P, int C) {
    size_t total = (size_t)P * C;
    size_t i = (size_t)blockIdx.x * 256 + threadIdx.x;
    if (i >= total) return;
    o[i] = (float)x[i] + ls[i % C] * (float)y[i];
}

// ------------------------------- host side ---------------------------------
namespace {
struct Arena { char* base; size_t off, cap; };
inline void* aalloc(Arena& a, size_t bytes) {
    size_t o = (a.off + 255) & ~(size_t)255;
    a.off = o + bytes;
    return a.base + o;
}
inline unsigned gsz(size_t n) { return (unsigned)((n + 255) / 256); }
}

extern "C" void kernel_launch(void* const* d_in, const int* in_sizes, int n_in,
                              void* d_out, int out_size, void* d_ws, size_t ws_size,
                              hipStream_t stream) {
    (void)in_sizes; (void)n_in; (void)out_size;
    Arena A{(char*)d_ws, 0, ws_size};
    auto alloc_h = [&](size_t n) { return (h16*)aalloc(A, n * sizeof(h16)); };
    auto alloc_f = [&](size_t n) { return (float*)aalloc(A, n * sizeof(float)); };
    auto Fp = [&](int i) { return (const float*)d_in[i]; };

    // ---- param leaf indices (jax pytree, dict keys sorted) ----
    const int P0 = 6;
    const int ATT = P0 + 0;                                      // afim.attn (26 leaves)
    const int LS1 = P0 + 26, LS2 = P0 + 27, MLPB = P0 + 28;      // ls1, ls2, mlp (8)
    const int AGG_O = P0 + 36, AGG_X = P0 + 84;                  // 48 each
    const int ORFB1 = P0 + 132, ORFB2 = P0 + 144, ORFB3 = P0 + 156;
    const int XRFB1 = P0 + 168, XRFB2 = P0 + 180, XRFB3 = P0 + 192;
    const int CC2 = 0, CC3 = 6, LAST = 12, UP1 = 18, UP2 = 24, UP3 = 30, UP4 = 36, UP5 = 42;

    const int Bz = 4;

    auto cbr = [&](int base, const h16* in, int N, int H, int W, int Ci, int Co) {
        h16* out = alloc_h((size_t)N * H * W * Co);
        size_t mk = A.off;
        size_t K = (size_t)9 * Ci;
        h16* wp = alloc_h(K * Co);
        float* sc = alloc_f(Co); float* bi = alloc_f(Co);
        k_pack_conv<<<gsz(K * Co), 256, 0, stream>>>(Fp(base + 5), wp, Ci, Co);
        k_bnfold<<<gsz(Co), 256, 0, stream>>>(Fp(base + 0), Fp(base + 1), Fp(base + 2),
                                              Fp(base + 3), Fp(base + 4), sc, bi, Co);
        dim3 g((N * H * W + 31) / 32, (Co + 63) / 64, 1);
        k_conv3x3<<<g, 128, 0, stream>>>(in, wp, sc, bi, out, N, H, W, Ci, Co);
        A.off = mk;                       // recycle packed weights (stream-ordered)
        return out;
    };
    auto gemm = [&](const h16* Ain, const float* Wf, const float* bias,
                    int M, int K, int N, int act) {
        h16* out = alloc_h((size_t)M * N);
        size_t mk = A.off;
        h16* wh = alloc_h((size_t)K * N);
        k_pack_lin<<<gsz((size_t)K * N), 256, 0, stream>>>(Wf, wh, K, N);
        dim3 g((M + 31) / 32, (N + 63) / 64, 1);
        k_gemm<<<g, 128, 0, stream>>>(Ain, wh, bias, out, nullptr, M, K, N, 0, 0, 0, act);
        A.off = mk;
        return out;
    };
    auto to_nhwc = [&](const float* in, int N, int C, int H, int W) {
        h16* o = alloc_h((size_t)N * C * H * W);
        k_nchw2nhwc<<<gsz((size_t)N * C * H * W), 256, 0, stream>>>(in, o, N, C, H, W);
        return o;
    };
    auto up2 = [&](const h16* in, int N, int H, int W, int C) {
        h16* o = alloc_h((size_t)N * 4 * H * W * C);
        k_up2<<<gsz((size_t)N * 4 * H * W * C), 256, 0, stream>>>(in, o, N, H, W, C);
        return o;
    };
    auto mul = [&](const h16* a, const h16* b, size_t n) {
        h16* o = alloc_h(n);
        k_mul<<<gsz(n), 256, 0, stream>>>(a, b, o, n);
        return o;
    };
    auto cat2 = [&](const h16* a, int Ca, const h16* b, int Cb, int P) {
        h16* o = alloc_h((size_t)P * (Ca + Cb));
        k_cat2<<<gsz((size_t)P * (Ca + Cb)), 256, 0, stream>>>(a, Ca, b, Cb, o, P);
        return o;
    };
    auto lnorm = [&](const h16* x, const float* w, const float* b, int P, int C) {
        h16* o = alloc_h((size_t)P * C);
        k_ln<<<gsz(P), 256, 0, stream>>>(x, w, b, o, P, C);
        return o;
    };
    auto dwc = [&](const h16* in, const float* w, const float* b,
                   int N, int H, int W, int C, int ks, int pad, int addres) {
        h16* o = alloc_h((size_t)N * H * W * C);
        k_dwconv<<<gsz((size_t)N * H * W * C), 256, 0, stream>>>(in, w, b, o, N, H, W, C, ks, pad, addres);
        return o;
    };
    auto run_ef = [&](int pb, const h16* in, int N, int H, int W, int Ci, int Co) {
        h16* t = cbr(pb + 0, in, N, H, W, Ci, Co);
        return cbr(pb + 6, t, N, H, W, Co, Co);
    };
    auto run_agg = [&](int pb, const h16* e4, const h16* e3, const h16* e2) {
        h16* u4 = up2(e4, Bz, 14, 14, 512);                                  // 512@28
        h16* e3_1 = mul(cbr(pb + UP1, u4, Bz, 28, 28, 512, 256), e3, (size_t)Bz * 28 * 28 * 256);
        h16* u44 = up2(u4, Bz, 28, 28, 512);                                 // 512@56
        h16* pa = cbr(pb + UP2, u44, Bz, 56, 56, 512, 128);
        h16* u3 = up2(e3, Bz, 28, 28, 256);                                  // 256@56
        h16* pb3 = cbr(pb + UP3, u3, Bz, 56, 56, 256, 128);
        h16* e2_1 = mul(mul(pa, pb3, (size_t)Bz * 56 * 56 * 128), e2, (size_t)Bz * 56 * 56 * 128);
        h16* p4 = cbr(pb + UP4, u4, Bz, 28, 28, 512, 512);
        h16* c1 = cat2(e3_1, 256, p4, 512, Bz * 28 * 28);                    // 768@28
        h16* e3_2 = cbr(pb + CC2, c1, Bz, 28, 28, 768, 768);
        h16* u5 = up2(e3_2, Bz, 28, 28, 768);                                // 768@56
        h16* p5 = cbr(pb + UP5, u5, Bz, 56, 56, 768, 768);
        h16* e2_2 = cat2(e2_1, 128, p5, 768, Bz * 56 * 56);                  // 896@56
        h16* c3 = cbr(pb + CC3, e2_2, Bz, 56, 56, 896, 896);
        return cbr(pb + LAST, c3, Bz, 56, 56, 896, 512);
    };

    const int HWp = 3136, P = Bz * HWp;               // 12544 tokens, C=512
    const size_t actB = (size_t)P * 512 * sizeof(h16);

    // persistent slots
    h16* oa = alloc_h((size_t)P * 512);
    h16* xa = alloc_h((size_t)P * 512);
    h16* x1 = alloc_h((size_t)P * 512);

    // ---- branch o: ef pyramid + agg ----
    {
        size_t mk = A.off;
        h16* o3c = to_nhwc(Fp(0), Bz, 512, 56, 56);
        h16* o4c = to_nhwc(Fp(1), Bz, 1024, 28, 28);
        h16* o5c = to_nhwc(Fp(2), Bz, 2048, 14, 14);
        h16* e2 = run_ef(ORFB1, o3c, Bz, 56, 56, 512, 128);
        h16* e3 = run_ef(ORFB2, o4c, Bz, 28, 28, 1024, 256);
        h16* e4 = run_ef(ORFB3, o5c, Bz, 14, 14, 2048, 512);
        h16* r = run_agg(AGG_O, e4, e3, e2);
        hipMemcpyAsync(oa, r, actB, hipMemcpyDeviceToDevice, stream);
        A.off = mk;
    }
    // ---- branch x ----
    {
        size_t mk = A.off;
        h16* x3c = to_nhwc(Fp(3), Bz, 512, 56, 56);
        h16* x4c = to_nhwc(Fp(4), Bz, 1024, 28, 28);
        h16* x5c = to_nhwc(Fp(5), Bz, 2048, 14, 14);
        h16* e2 = run_ef(XRFB1, x3c, Bz, 56, 56, 512, 128);
        h16* e3 = run_ef(XRFB2, x4c, Bz, 28, 28, 1024, 256);
        h16* e4 = run_ef(XRFB3, x5c, Bz, 14, 14, 2048, 512);
        h16* r = run_agg(AGG_X, e4, e3, e2);
        hipMemcpyAsync(xa, r, actB, hipMemcpyDeviceToDevice, stream);
        A.off = mk;
    }

    // ---- AFIM block: attention ----
    {
        size_t mk = A.off;
        h16* xn = lnorm(oa, Fp(ATT + 17), Fp(ATT + 14), P, 512);   // nw, nb
        h16* xen = lnorm(xa, Fp(ATT + 16), Fp(ATT + 15), P, 512);  // new, neb
        h16* q = gemm(xn, Fp(ATT + 21), Fp(ATT + 20), P, 512, 512, 0);
        h16* cut = gemm(xn, Fp(ATT + 23), Fp(ATT + 22), P, 512, 512, 0);
        h16* l = gemm(xn, Fp(ATT + 13), Fp(ATT + 12), P, 512, 512, 2);      // + GELU
        h16* lc = dwc(l, Fp(ATT + 3), Fp(ATT + 2), Bz, 56, 56, 512, 7, 3, 0);
        h16* aab = gemm(lc, Fp(ATT + 1), Fp(ATT + 0), P, 512, 512, 0);
        h16* kvb = gemm(l, Fp(ATT + 11), Fp(ATT + 10), P, 512, 512, 0);
        h16* kT = alloc_h((size_t)32 * 3136 * 32);
        h16* vv = alloc_h((size_t)32 * 3136 * 32);
        k_split_kv<<<gsz((size_t)P * 512), 256, 0, stream>>>(kvb, kT, vv, Bz);
        h16* mt = alloc_h((size_t)32 * 98 * 32);
        {
            size_t mk2 = A.off;
            h16* sc = cat2(xn, 512, xen, 512, P);                           // B,56,56,1024
            h16* pooled = alloc_h((size_t)Bz * 49 * 1024);
            k_pool8<<<gsz((size_t)Bz * 49 * 1024), 256, 0, stream>>>(sc, pooled, Bz, 1024);
            h16* pm = gemm(pooled, Fp(ATT + 25), Fp(ATT + 24), Bz * 49, 1024, 512, 0);
            k_build_m<<<gsz((size_t)Bz * 49 * 512), 256, 0, stream>>>(pm, mt, Bz);
            A.off = mk2;
        }
        // scores = m @ k^T (batched 32 x [98,32]@[32,3136]) -> softmax -> @ v
        h16* att = alloc_h((size_t)32 * 98 * 3136);
        {
            size_t mk2 = A.off;
            float* scores = alloc_f((size_t)32 * 98 * 3136);
            dim3 g1((98 + 31) / 32, 3136 / 64, 32);
            k_gemm<<<g1, 128, 0, stream>>>(mt, kT, nullptr, nullptr, scores,
                                           98, 32, 3136, 98 * 32, 3136 * 32, (long long)98 * 3136, 0);
            k_softmax<<<32 * 98, 256, 0, stream>>>(scores, att, 3136);
            A.off = mk2;
        }
        h16* ov = alloc_h((size_t)32 * 98 * 32);
        dim3 g2((98 + 31) / 32, 1, 32);
        k_gemm<<<g2, 128, 0, stream>>>(att, vv, nullptr, ov, nullptr,
                                       98, 3136, 32, (long long)98 * 3136, 3136 * 32, 98 * 32, 0);
        h16* oup = alloc_h((size_t)P * 512);
        k_o_up<<<gsz((size_t)P * 512), 256, 0, stream>>>(ov, oup, Bz);
        // edge branch
        h16* ef1 = gemm(xen, Fp(ATT + 7), Fp(ATT + 6), P, 512, 512, 0);     // e_fore
        h16* ecv = dwc(ef1, Fp(ATT + 9), Fp(ATT + 8), Bz, 56, 56, 512, 7, 3, 0);
        h16* xe2 = gemm(ecv, Fp(ATT + 5), Fp(ATT + 4), P, 512, 512, 0);     // e_back
        // concat + proj
        h16* t1 = mul(q, aab, (size_t)P * 512);
        h16* t2 = mul(cut, xe2, (size_t)P * 512);
        h16* cat = alloc_h((size_t)P * 1536);
        k_cat3<<<gsz((size_t)P * 1536), 256, 0, stream>>>(t1, oup, t2, cat, P, 512);
        h16* attn_out = gemm(cat, Fp(ATT + 19), Fp(ATT + 18), P, 1536, 512, 0);
        k_axpy<<<gsz((size_t)P * 512), 256, 0, stream>>>(oa, attn_out, Fp(LS1), x1, P, 512);
        A.off = mk;
    }

    // ---- AFIM block: MLP -> d_out (f32 NHWC) ----
    {
        size_t mk = A.off;
        h16* mn = lnorm(x1, Fp(MLPB + 5), Fp(MLPB + 4), P, 512);            // nw, nb
        h16* hh = gemm(mn, Fp(MLPB + 1), Fp(MLPB + 0), P, 512, 2048, 0);    // fc1
        h16* hp = dwc(hh, Fp(MLPB + 7), Fp(MLPB + 6), Bz, 56, 56, 2048, 3, 1, 1);  // pos + residual
        h16* hg = alloc_h((size_t)P * 2048);
        k_gelu<<<gsz((size_t)P * 2048), 256, 0, stream>>>(hp, hg, (size_t)P * 2048);
        h16* m2 = gemm(hg, Fp(MLPB + 3), Fp(MLPB + 2), P, 2048, 512, 0);    // fc2
        k_axpy_f32<<<gsz((size_t)P * 512), 256, 0, stream>>>(x1, m2, Fp(LS2), (float*)d_out, P, 512);
        A.off = mk;
    }
}
